// GroupedQueryAttention_42528766165180
// MI455X (gfx1250) — matrix operand order repaired and checked by reading, not compiled, not run on hardware
//
#include <hip/hip_runtime.h>

typedef __bf16 bf16_t;
typedef __attribute__((ext_vector_type(16))) __bf16 v16bf;
typedef __attribute__((ext_vector_type(8)))  __bf16 v8bf;
typedef __attribute__((ext_vector_type(8)))  float  v8f;
typedef __attribute__((ext_vector_type(4)))  int    v4i;

static constexpr int Bb = 2, Ss = 2048, DIN = 2048, Hh = 16, Gg = 4, Dd = 128;
static constexpr int MROWS = Bb * Ss;     // 4096
static constexpr int NQ  = Hh * Dd;       // 2048
static constexpr int NKV = Gg * Dd;       // 512

// Use CDNA5 async global->LDS copies when the toolchain exposes them.
#if defined(__HIP_DEVICE_COMPILE__)
#if __has_builtin(__builtin_amdgcn_global_load_async_to_lds_b128) && \
    __has_builtin(__builtin_amdgcn_s_wait_asynccnt)
#define USE_ASYNC_LDS 1
#endif
#endif
#ifndef USE_ASYNC_LDS
#define USE_ASYNC_LDS 0
#endif

#if USE_ASYNC_LDS
typedef __attribute__((address_space(1))) v4i g_v4i;   // global 16B chunk
typedef __attribute__((address_space(3))) v4i l_v4i;   // LDS 16B chunk
// 16B per lane, ASYNCcnt-tracked, no VGPR round trip.
#define ASYNC_CP_B128(gsrc, ldst)                                   \
  __builtin_amdgcn_global_load_async_to_lds_b128(                   \
      (g_v4i*)(gsrc), (l_v4i*)(ldst), 0, 0)
#define ASYNC_WAIT() __builtin_amdgcn_s_wait_asynccnt(0)
#else
#define ASYNC_WAIT()
#endif

// A-fragment (16x32 bf16): lane holds row M=lane%16; halves 0..7 = K(base..base+7),
// halves 8..15 = K(base+16..base+23), base = (lane<16 ? 0 : 8).  Two 16B loads.
__device__ __forceinline__ v16bf ld_frag_gap(const bf16_t* p) {
  union { v16bf v; v8bf h[2]; } u;
  u.h[0] = *(const v8bf*)(p);
  u.h[1] = *(const v8bf*)(p + 16);
  return u.v;
}

__device__ __forceinline__ v8f wmma_bf16(v16bf a, v16bf b, v8f c) {
  return __builtin_amdgcn_wmma_f32_16x16x32_bf16(false, a, false, b, (short)0, c,
                                                 false, false);
}

// ---------------------------------------------------------------- fp32 -> bf16
__global__ __launch_bounds__(256) void cvt_f32_bf16(const float* __restrict__ in,
                                                    bf16_t* __restrict__ out, int n) {
  int i = blockIdx.x * blockDim.x + threadIdx.x;
  int stride = gridDim.x * blockDim.x;
  for (; i < n; i += stride) out[i] = (bf16_t)in[i];
}

// -------------------------------------------------- 128x128 tiled bf16 WMMA GEMM
// C[M,N] = A[M,K] @ W[K,N]; A,W bf16 row-major, accumulate fp32.
// Double-buffered LDS; A tile staged via async global->LDS when available.
template <int OUT_BF16>
__global__ __launch_bounds__(256) void gemm_bf16_128x128(
    const bf16_t* __restrict__ A, const bf16_t* __restrict__ W,
    void* __restrict__ C, int M, int N, int K) {
  __shared__ bf16_t sA[2][128 * 32];   // [m][k]
  __shared__ bf16_t sB[2][128 * 32];   // transposed: [n][k]
  const int t = threadIdx.x;
  const int wave = t >> 5, lane = t & 31;
  const int half = lane >> 4, l16 = lane & 15;
  const int m0w = (wave >> 1) * 32, n0w = (wave & 1) * 64;   // 4x2 wave grid
  const int bm = blockIdx.y * 128, bn = blockIdx.x * 128;

  v8f acc[2][4];
#pragma unroll
  for (int i = 0; i < 2; i++)
#pragma unroll
    for (int j = 0; j < 4; j++)
#pragma unroll
      for (int e = 0; e < 8; e++) acc[i][j][e] = 0.0f;

  const int arow = t >> 1, ak = (t & 1) * 16;     // A stage: 128 rows x 32
  const int brow = t >> 3, bno = (t & 7) * 16;    // W stage: 32 rows x 128

  auto stageA = [&](int buf, int k0) {
    const bf16_t* src = A + (size_t)(bm + arow) * K + k0 + ak;
    bf16_t* dst = &sA[buf][arow * 32 + ak];
#if USE_ASYNC_LDS
    ASYNC_CP_B128(src, dst);
    ASYNC_CP_B128(src + 8, dst + 8);
#else
    *(v16bf*)dst = *(const v16bf*)src;
#endif
  };
  auto stageB = [&](int buf, int k0) {   // transpose 32x128 -> [n][k]
    v16bf wvec = *(const v16bf*)(W + (size_t)(k0 + brow) * N + bn + bno);
#pragma unroll
    for (int j = 0; j < 16; j++) sB[buf][(bno + j) * 32 + brow] = wvec[j];
  };

  const int nk = K / 32;
  stageA(0, 0);
  stageB(0, 0);
  ASYNC_WAIT();
  __syncthreads();

  for (int it = 0; it < nk; it++) {
    const int cur = it & 1;
    if (it + 1 < nk) {                       // stage next tile into other buffer
      stageA(cur ^ 1, (it + 1) * 32);
      stageB(cur ^ 1, (it + 1) * 32);
    }
    if (it + 2 < nk) {                       // warm L2 two tiles ahead
      __builtin_prefetch(A + (size_t)(bm + arow) * K + (it + 2) * 32 + ak, 0, 1);
      __builtin_prefetch(W + (size_t)((it + 2) * 32 + brow) * N + bn + bno, 0, 1);
    }

    v16bf afr[2], bfr[4];
#pragma unroll
    for (int mi = 0; mi < 2; mi++)
      afr[mi] = ld_frag_gap(&sA[cur][(m0w + mi * 16 + l16) * 32 + half * 8]);
#pragma unroll
    for (int ni = 0; ni < 4; ni++)   // B-frag: lane col n, K = half*8..+7, half*8+16..+23
      bfr[ni] = ld_frag_gap(&sB[cur][(n0w + ni * 16 + l16) * 32 + half * 8]);
#pragma unroll
    for (int mi = 0; mi < 2; mi++)
#pragma unroll
      for (int ni = 0; ni < 4; ni++)
        acc[mi][ni] = wmma_bf16(afr[mi], bfr[ni], acc[mi][ni]);

    ASYNC_WAIT();
    __syncthreads();
  }

#pragma unroll
  for (int mi = 0; mi < 2; mi++)
#pragma unroll
    for (int ni = 0; ni < 4; ni++)
#pragma unroll
      for (int i = 0; i < 8; i++) {
        int r = bm + m0w + mi * 16 + half * 8 + i;   // C layout: rows half*8+i
        int c = bn + n0w + ni * 16 + l16;            // col = lane%16
        float v = acc[mi][ni][i];
        if (OUT_BF16) ((bf16_t*)C)[(size_t)r * N + c] = (bf16_t)v;
        else          ((float*)C)[(size_t)r * N + c] = v;
      }
}

// ----------------------------------------------- fused RMSNorm + RoPE, fp32->bf16
// Rows of 128 contiguous values per (b,s,head); one wave per row.
__global__ __launch_bounds__(256) void rmsnorm_rope(
    const float* __restrict__ in, bf16_t* __restrict__ out,
    const float* __restrict__ w, const float* __restrict__ cosb,
    const float* __restrict__ sinb, int nrows, int nheads, float outscale) {
  int row = blockIdx.x * 8 + (threadIdx.x >> 5);
  if (row >= nrows) return;
  int lane = threadIdx.x & 31;
  int s = (row / nheads) % Ss;
  const float* x = in + (size_t)row * Dd;
  int d0 = lane * 4;
  float v[4], p[4];
  float ss = 0.f;
#pragma unroll
  for (int j = 0; j < 4; j++) { v[j] = x[d0 + j]; ss += v[j] * v[j]; }
#pragma unroll
  for (int m = 1; m < 32; m <<= 1) ss += __shfl_xor(ss, m, 32);
  float rs = rsqrtf(ss * (1.0f / Dd) + 1e-6f);
  int pd0 = (d0 < 64) ? d0 + 64 : d0 - 64;
#pragma unroll
  for (int j = 0; j < 4; j++) p[j] = x[pd0 + j];
#pragma unroll
  for (int j = 0; j < 4; j++) {
    int d = d0 + j;
    float xn = v[j] * rs * w[d];
    float pn = p[j] * rs * w[pd0 + j];
    float rot = (d < 64) ? -pn : pn;
    float o = xn * cosb[s * Dd + d] + rot * sinb[s * Dd + d];
    out[(size_t)row * Dd + d] = (bf16_t)(o * outscale);
  }
}

// ------------------------------------------------------- flash attention (causal)
// block = (q-tile of 128, head h, batch b); 8 waves x 16 q-rows each.
// V chunk staging double-buffered against compute.
__global__ __launch_bounds__(256) void attn_fwd(
    const bf16_t* __restrict__ Q, const bf16_t* __restrict__ Kb,
    const bf16_t* __restrict__ Vb, bf16_t* __restrict__ O) {
  __shared__ bf16_t sVt[2][Dd * 32];   // V transposed: [d][key], 2 x 8 KB
  __shared__ bf16_t sP[8 * 16 * 32];   // per-wave P tile, 8 KB
  const int t = threadIdx.x;
  const int wv = t >> 5, lane = t & 31, half = lane >> 4, l16 = lane & 15;
  const int q0 = blockIdx.x * 128;
  const int h = blockIdx.y, b = blockIdx.z;
  const int g = h / (Hh / Gg);
  const int qw0 = q0 + wv * 16;

  // preload Q A-fragments for all 4 K-steps of D=128 (Q already scaled by D^-1/2)
  const bf16_t* qbase =
      Q + ((size_t)(b * Ss + qw0 + l16)) * NQ + h * Dd + half * 8;
  v16bf qfr[4];
#pragma unroll
  for (int ds = 0; ds < 4; ds++) qfr[ds] = ld_frag_gap(qbase + ds * 32);

  v8f ctx[8];
#pragma unroll
  for (int i = 0; i < 8; i++)
#pragma unroll
    for (int e = 0; e < 8; e++) ctx[i][e] = 0.0f;
  float mrow[8], lrow[8];
#pragma unroll
  for (int i = 0; i < 8; i++) { mrow[i] = -__builtin_inff(); lrow[i] = 0.f; }

  const int nchunk = q0 / 32 + 4;   // causal: keys <= q0+127
  bf16_t* myP = sP + wv * (16 * 32);

  auto stageV = [&](int buf, int kc) {   // transpose 32 keys x 128 d -> [d][key]
    int key = kc * 32 + (t >> 3);
    int dd0 = (t & 7) * 16;
    v16bf vv =
        *(const v16bf*)(Vb + ((size_t)(b * Ss + key)) * NKV + g * Dd + dd0);
#pragma unroll
    for (int j = 0; j < 16; j++) sVt[buf][(dd0 + j) * 32 + (t >> 3)] = vv[j];
  };

  stageV(0, 0);
  __syncthreads();

  for (int kc = 0; kc < nchunk; kc++) {
    const int cur = kc & 1;
    const int kbase0 = kc * 32;
    if (kc + 1 < nchunk) stageV(cur ^ 1, kc + 1);

    if (kbase0 <= qw0 + 15) {   // wave-uniform: chunk intersects causal window
      v8f sc[2];
#pragma unroll
      for (int t2 = 0; t2 < 2; t2++) {
        const int kk0 = kbase0 + t2 * 16;
        v8f a;
#pragma unroll
        for (int e = 0; e < 8; e++) a[e] = 0.0f;
#pragma unroll
        for (int ds = 0; ds < 4; ds++) {
          // K B-fragment straight from global: lane=key col, d's half*8..+7, half*8+16..+23
          v16bf kf = ld_frag_gap(Kb + ((size_t)(b * Ss + kk0 + l16)) * NKV +
                                     g * Dd + ds * 32 + half * 8);
          a = wmma_bf16(qfr[ds], kf, a);
        }
#pragma unroll
        for (int i = 0; i < 8; i++) {   // causal mask
          int r = qw0 + half * 8 + i;
          if (kk0 + l16 > r) a[i] = -__builtin_inff();
        }
        sc[t2] = a;
      }
      // online softmax over the 32-key chunk
#pragma unroll
      for (int i = 0; i < 8; i++) {
        float mx = fmaxf(sc[0][i], sc[1][i]);
#pragma unroll
        for (int m = 1; m < 16; m <<= 1) mx = fmaxf(mx, __shfl_xor(mx, m, 32));
        float mnew = fmaxf(mrow[i], mx);
        float scale = __expf(mrow[i] - mnew);
        float p0 = __expf(sc[0][i] - mnew);
        float p1 = __expf(sc[1][i] - mnew);
        sc[0][i] = p0; sc[1][i] = p1;
        float sum = p0 + p1;
#pragma unroll
        for (int m = 1; m < 16; m <<= 1) sum += __shfl_xor(sum, m, 32);
        lrow[i] = lrow[i] * scale + sum;
        mrow[i] = mnew;
#pragma unroll
        for (int dt = 0; dt < 8; dt++) ctx[dt][i] *= scale;
      }
      // reshape P: C-fragment -> A-fragment via per-wave LDS slot
#pragma unroll
      for (int t2 = 0; t2 < 2; t2++)
#pragma unroll
        for (int i = 0; i < 8; i++)
          myP[(half * 8 + i) * 32 + t2 * 16 + l16] = (bf16_t)sc[t2][i];
      v16bf pfr = ld_frag_gap(myP + l16 * 32 + half * 8);
#pragma unroll
      for (int dt = 0; dt < 8; dt++) {
        v16bf vf = ld_frag_gap(&sVt[cur][(dt * 16 + l16) * 32 + half * 8]);
        ctx[dt] = wmma_bf16(pfr, vf, ctx[dt]);
      }
    }
    __syncthreads();
  }

#pragma unroll
  for (int dt = 0; dt < 8; dt++)
#pragma unroll
    for (int i = 0; i < 8; i++) {
      int r = qw0 + half * 8 + i;
      float o = ctx[dt][i] / lrow[i];
      O[((size_t)(b * Ss + r)) * NQ + h * Dd + dt * 16 + l16] = (bf16_t)o;
    }
}

// --------------------------------------------------------------------- launcher
extern "C" void kernel_launch(void* const* d_in, const int* in_sizes, int n_in,
                              void* d_out, int out_size, void* d_ws, size_t ws_size,
                              hipStream_t stream) {
  (void)in_sizes; (void)n_in; (void)out_size; (void)ws_size;
  const float* x    = (const float*)d_in[0];
  // d_in[1] = mask: causal triu, implemented analytically in attn_fwd
  const float* cosb = (const float*)d_in[2];
  const float* sinb = (const float*)d_in[3];
  const float* Wq   = (const float*)d_in[4];
  const float* Wk   = (const float*)d_in[5];
  const float* Wv   = (const float*)d_in[6];
  const float* Wo   = (const float*)d_in[7];
  const float* qw   = (const float*)d_in[8];
  const float* kw   = (const float*)d_in[9];
  float* out = (float*)d_out;

  char* ws = (char*)d_ws;
  size_t off = 0;
  auto alloc = [&](size_t bytes) {
    char* p = ws + off;
    off += (bytes + 255) & ~(size_t)255;
    return p;
  };
  bf16_t* xh  = (bf16_t*)alloc((size_t)MROWS * DIN * 2);   // 16 MB
  bf16_t* wqh = (bf16_t*)alloc((size_t)DIN * NQ * 2);      //  8 MB
  bf16_t* wkh = (bf16_t*)alloc((size_t)DIN * NKV * 2);     //  2 MB
  bf16_t* wvh = (bf16_t*)alloc((size_t)DIN * NKV * 2);     //  2 MB
  bf16_t* woh = (bf16_t*)alloc((size_t)NQ * DIN * 2);      //  8 MB
  float*  qf  = (float*) alloc((size_t)MROWS * NQ * 4);    // 32 MB
  float*  kf  = (float*) alloc((size_t)MROWS * NKV * 4);   //  8 MB
  bf16_t* vh  = (bf16_t*)alloc((size_t)MROWS * NKV * 2);   //  4 MB
  bf16_t* qh  = (bf16_t*)alloc((size_t)MROWS * NQ * 2);    // 16 MB
  bf16_t* kh  = (bf16_t*)alloc((size_t)MROWS * NKV * 2);   //  4 MB
  bf16_t* ch  = (bf16_t*)alloc((size_t)MROWS * NQ * 2);    // 16 MB

  auto cvt = [&](const float* src, bf16_t* dst, int n) {
    int blocks = (n + 1023) / 1024; if (blocks > 4096) blocks = 4096;
    cvt_f32_bf16<<<blocks, 256, 0, stream>>>(src, dst, n);
  };
  cvt(x,  xh,  MROWS * DIN);
  cvt(Wq, wqh, DIN * NQ);
  cvt(Wk, wkh, DIN * NKV);
  cvt(Wv, wvh, DIN * NKV);
  cvt(Wo, woh, NQ * DIN);

  dim3 blk(256);
  gemm_bf16_128x128<0><<<dim3(NQ / 128, MROWS / 128), blk, 0, stream>>>(
      xh, wqh, qf, MROWS, NQ, DIN);
  gemm_bf16_128x128<0><<<dim3(NKV / 128, MROWS / 128), blk, 0, stream>>>(
      xh, wkh, kf, MROWS, NKV, DIN);
  gemm_bf16_128x128<1><<<dim3(NKV / 128, MROWS / 128), blk, 0, stream>>>(
      xh, wvh, vh, MROWS, NKV, DIN);

  // q: fold D^-1/2 scaling into the bf16 q tensor
  rmsnorm_rope<<<(MROWS * Hh) / 8, 256, 0, stream>>>(
      qf, qh, qw, cosb, sinb, MROWS * Hh, Hh, 0.08838834764831845f);
  rmsnorm_rope<<<(MROWS * Gg) / 8, 256, 0, stream>>>(
      kf, kh, kw, cosb, sinb, MROWS * Gg, Gg, 1.0f);

  attn_fwd<<<dim3(Ss / 128, Hh, Bb), blk, 0, stream>>>(qh, kh, vh, ch);

  gemm_bf16_128x128<0><<<dim3(DIN / 128, MROWS / 128), blk, 0, stream>>>(
      ch, woh, out, MROWS, DIN, NQ);
}
